// Network_88699664597471
// MI455X (gfx1250) — compile-verified, matrix operand order checked by reading, not run on hardware
//
#include <hip/hip_runtime.h>
#include <math.h>

typedef __attribute__((ext_vector_type(16))) _Float16 v16h;
typedef __attribute__((ext_vector_type(8)))  float    v8f;

// ---------------------------------------------------------------------------
// WMMA 16x16 tile GEMM core:  D = A(16xK) * B(Kx16),  B[k][n] = W[n][k]
// A: row-major [M x Kpad] f16, W: row-major [O x Kpad] f16, Kpad % 32 == 0.
// Fragment layouts per CDNA5 ISA 7.12.2 (wave32).
// ---------------------------------------------------------------------------
__device__ __forceinline__ v8f wmma_mm(const _Float16* __restrict__ A,
                                       const _Float16* __restrict__ W,
                                       int m0, int n0, int Kpad) {
  const int l  = threadIdx.x;
  const int s  = l >> 4;       // lane half
  const int lm = l & 15;
  const unsigned* arow = (const unsigned*)(A + (size_t)(m0 + lm) * Kpad);
  const unsigned* brow = (const unsigned*)(W + (size_t)(n0 + lm) * Kpad);
  v8f acc = {};
  for (int k0 = 0; k0 < Kpad; k0 += 32) {
    union HU { v16h v; unsigned u[8]; } ua, ub;
#pragma unroll
    for (int p = 0; p < 8; p++) {
      const int k = k0 + ((p < 4) ? (2 * p) : (16 + 2 * (p - 4))) + 8 * s;
      ua.u[p] = arow[k >> 1];
      ub.u[p] = brow[k >> 1];
    }
    acc = __builtin_amdgcn_wmma_f32_16x16x32_f16(false, ua.v, false, ub.v,
                                                 (short)0, acc, false, false);
  }
  return acc;
}

// Pass 1: GEMM, accumulate per-channel sum / sumsq (BN batch statistics).
__global__ __launch_bounds__(32) void k_gemm_stats(
    const _Float16* __restrict__ A, const _Float16* __restrict__ W,
    int Kpad, int O, float* __restrict__ stats) {
  const int m0 = blockIdx.x * 16, n0 = blockIdx.y * 16;
  v8f acc = wmma_mm(A, W, m0, n0, Kpad);
  const int lm = threadIdx.x & 15;
  float su = 0.f, sq = 0.f;
#pragma unroll
  for (int r = 0; r < 8; r++) { float v = acc[r]; su += v; sq += v * v; }
  atomicAdd(&stats[n0 + lm], su);
  atomicAdd(&stats[O + n0 + lm], sq);
}

// Pass 2: recompute GEMM, fused BN+ReLU epilogue.
//  mode 0: write f16 activations for next layer (stride O)
//  mode 1: atomicMax-pool over `aux` consecutive rows into f32 fout (SA tail)
//  mode 2: write f32 features (FP tail)
//  mode 3: write f32 only rows where m % aux == 0 (FC -> net1)
__global__ __launch_bounds__(32) void k_gemm_apply(
    const _Float16* __restrict__ A, const _Float16* __restrict__ W,
    int Kpad, int O, const float* __restrict__ ss, int mode,
    _Float16* __restrict__ Anext, float* __restrict__ fout, int aux) {
  const int m0 = blockIdx.x * 16, n0 = blockIdx.y * 16;
  v8f acc = wmma_mm(A, W, m0, n0, Kpad);
  const int l = threadIdx.x, s = l >> 4, lm = l & 15;
  const int n = n0 + lm;
  const float sc = ss[n], sh = ss[O + n];
#pragma unroll
  for (int r = 0; r < 8; r++) {
    float v = fmaxf(fmaf(acc[r], sc, sh), 0.0f);
    int m = m0 + r + 8 * s;
    if (mode == 0) {
      Anext[(size_t)m * O + n] = (_Float16)v;
    } else if (mode == 1) {
      atomicMax((int*)&fout[(size_t)(m / aux) * O + n], __float_as_int(v));
    } else if (mode == 2) {
      fout[(size_t)m * O + n] = v;
    } else {
      if ((m % aux) == 0) fout[(size_t)(m / aux) * O + n] = v;
    }
  }
}

__global__ void k_finalize(const float* __restrict__ stats,
                           const float* __restrict__ gamma,
                           const float* __restrict__ beta,
                           float minv, int O, float* __restrict__ ss) {
  int o = threadIdx.x;
  if (o >= O) return;
  float mu  = stats[o] * minv;
  float var = stats[O + o] * minv - mu * mu;
  float sc  = gamma[o] * rsqrtf(var + 1e-5f);
  ss[o] = sc;
  ss[O + o] = beta[o] - mu * sc;
}

__global__ void k_zero_f(float* p, size_t n) {
  size_t i = (size_t)blockIdx.x * blockDim.x + threadIdx.x;
  size_t st = (size_t)gridDim.x * blockDim.x;
  for (; i < n; i += st) p[i] = 0.0f;
}

__global__ void k_wcvt(const float* __restrict__ W, _Float16* __restrict__ Wh,
                       int K, int Kpad, int total) {
  int t = blockIdx.x * blockDim.x + threadIdx.x;
  if (t >= total) return;
  int o = t / Kpad, k = t % Kpad;
  Wh[t] = (_Float16)((k < K) ? W[o * K + k] : 0.0f);
}

__global__ void k_cast_f2h(const float* __restrict__ s, _Float16* __restrict__ d,
                           size_t n) {
  size_t i = (size_t)blockIdx.x * blockDim.x + threadIdx.x;
  size_t st = (size_t)gridDim.x * blockDim.x;
  for (; i < n; i += st) d[i] = (_Float16)s[i];
}

// pcs with row 0 replaced by cp1
__global__ void k_init_pcs(const float* __restrict__ pcs,
                           const float* __restrict__ cp1,
                           float* __restrict__ out, int N, int total) {
  int t = blockIdx.x * blockDim.x + threadIdx.x;
  if (t >= total) return;
  int c = t % 3, r = t / 3;
  int pt = r % N, b = r / N;
  out[t] = (pt == 0) ? cp1[b * 3 + c] : pcs[t];
}

// Farthest point sampling: one workgroup per batch, dist[] resident in LDS.
__global__ void k_fps(const float* __restrict__ xyz, int N, int np,
                      int* __restrict__ out) {
  __shared__ float dist[4096];
  __shared__ float rv[256];
  __shared__ int   ri[256];
  const int b = blockIdx.x, t = threadIdx.x;
  const float* p = xyz + (size_t)b * N * 3;
  for (int j = t; j < N; j += 256) dist[j] = 1e10f;
  __syncthreads();
  int last = 0;
  for (int i = 0; i < np; i++) {
    if (t == 0) out[b * np + i] = last;
    float cx = p[last * 3], cy = p[last * 3 + 1], cz = p[last * 3 + 2];
    float bv = -1.0f; int bi = 0;
    for (int j = t; j < N; j += 256) {
      float dx = p[j * 3] - cx, dy = p[j * 3 + 1] - cy, dz = p[j * 3 + 2] - cz;
      float nd = fminf(dist[j], dx * dx + dy * dy + dz * dz);
      dist[j] = nd;
      if (nd > bv) { bv = nd; bi = j; }
    }
    rv[t] = bv; ri[t] = bi;
    __syncthreads();
    for (int off = 128; off > 0; off >>= 1) {
      if (t < off) {
        if (rv[t + off] > rv[t] ||
            (rv[t + off] == rv[t] && ri[t + off] < ri[t])) {
          rv[t] = rv[t + off]; ri[t] = ri[t + off];
        }
      }
      __syncthreads();
    }
    last = ri[0];
    __syncthreads();
  }
}

__global__ void k_gather_xyz(const float* __restrict__ src,
                             const int* __restrict__ idx,
                             float* __restrict__ dst, int np, int Nsrc,
                             int total) {
  int t = blockIdx.x * blockDim.x + threadIdx.x;
  if (t >= total) return;
  int c = t % 3, r = t / 3;
  int pt = r % np, b = r / np;
  dst[t] = src[((size_t)b * Nsrc + idx[b * np + pt]) * 3 + c];
}

// Ball query: first `ns` in-radius indices in ascending order, padded w/ first.
__global__ void k_ball(const float* __restrict__ src,
                       const float* __restrict__ q, int Nsrc, int np, int ns,
                       float r2, int* __restrict__ out, int total) {
  int t = blockIdx.x * blockDim.x + threadIdx.x;
  if (t >= total) return;
  int b = t / np, pt = t % np;
  const float* S = src + (size_t)b * Nsrc * 3;
  float cx = q[(size_t)t * 3], cy = q[(size_t)t * 3 + 1], cz = q[(size_t)t * 3 + 2];
  int* o = out + (size_t)t * ns;
  int cnt = 0, first = Nsrc - 1;
  for (int j = 0; j < Nsrc && cnt < ns; j++) {
    float dx = S[j * 3] - cx, dy = S[j * 3 + 1] - cy, dz = S[j * 3 + 2] - cz;
    if (dx * dx + dy * dy + dz * dz < r2) {
      if (cnt == 0) first = j;
      o[cnt++] = j;
    }
  }
  for (int k = cnt; k < ns; k++) o[k] = first;
}

// Build grouped SA input: concat(centered xyz, feats) -> f16, K zero-padded.
__global__ void k_group(const float* __restrict__ sxyz,
                        const float* __restrict__ sfeat,
                        const float* __restrict__ q,
                        const int* __restrict__ bidx, int Nsrc, int np, int ns,
                        int F, int Kpad, _Float16* __restrict__ A, int total) {
  int t = blockIdx.x * blockDim.x + threadIdx.x;
  if (t >= total) return;
  int c = t % Kpad; int r = t / Kpad;
  int smp = r % ns; int r2 = r / ns;
  int pt = r2 % np; int b = r2 / np;
  int id = bidx[r];
  float v = 0.0f;
  if (c < 3)
    v = sxyz[((size_t)b * Nsrc + id) * 3 + c] - q[((size_t)b * np + pt) * 3 + c];
  else if (c < 3 + F)
    v = sfeat[((size_t)b * Nsrc + id) * F + (c - 3)];
  A[(size_t)r * Kpad + c] = (_Float16)v;
}

// 3-NN + normalized inverse-distance weights.
__global__ void k_3nn(const float* __restrict__ x1, const float* __restrict__ x2,
                      int N1, int N2, int* __restrict__ nidx,
                      float* __restrict__ nw, int total) {
  int t = blockIdx.x * blockDim.x + threadIdx.x;
  if (t >= total) return;
  int b = t / N1;
  const float* P = x2 + (size_t)b * N2 * 3;
  float px = x1[(size_t)t * 3], py = x1[(size_t)t * 3 + 1], pz = x1[(size_t)t * 3 + 2];
  float b0 = 1e30f, b1 = 1e30f, b2 = 1e30f;
  int i0 = 0, i1 = 0, i2 = 0;
  for (int j = 0; j < N2; j++) {
    float dx = P[j * 3] - px, dy = P[j * 3 + 1] - py, dz = P[j * 3 + 2] - pz;
    float d = dx * dx + dy * dy + dz * dz;
    if (d < b0)      { b2 = b1; i2 = i1; b1 = b0; i1 = i0; b0 = d; i0 = j; }
    else if (d < b1) { b2 = b1; i2 = i1; b1 = d; i1 = j; }
    else if (d < b2) { b2 = d; i2 = j; }
  }
  float w0 = 1.0f / (b0 + 1e-8f), w1 = 1.0f / (b1 + 1e-8f), w2 = 1.0f / (b2 + 1e-8f);
  float inv = 1.0f / (w0 + w1 + w2);
  nidx[t * 3 + 0] = i0; nidx[t * 3 + 1] = i1; nidx[t * 3 + 2] = i2;
  nw[t * 3 + 0] = w0 * inv; nw[t * 3 + 1] = w1 * inv; nw[t * 3 + 2] = w2 * inv;
}

// Build FP input: concat(interp(f2), f1) -> f16, K zero-padded.
__global__ void k_fp_build(const float* __restrict__ f2,
                           const float* __restrict__ f1,
                           const int* __restrict__ nidx,
                           const float* __restrict__ nw, int np, int N2, int F2,
                           int F1, int Kpad, _Float16* __restrict__ A,
                           int total) {
  int t = blockIdx.x * blockDim.x + threadIdx.x;
  if (t >= total) return;
  int c = t % Kpad; int r = t / Kpad;
  int b = r / np;
  float v = 0.0f;
  if (c < F2) {
    const float* Fb = f2 + (size_t)b * N2 * F2;
    v = nw[r * 3 + 0] * Fb[(size_t)nidx[r * 3 + 0] * F2 + c] +
        nw[r * 3 + 1] * Fb[(size_t)nidx[r * 3 + 1] * F2 + c] +
        nw[r * 3 + 2] * Fb[(size_t)nidx[r * 3 + 2] * F2 + c];
  } else if (c < F2 + F1) {
    v = f1[(size_t)r * F1 + (c - F2)];
  }
  A[(size_t)r * Kpad + c] = (_Float16)v;
}

// Heads: task/cp linear -> concat -> leaky_relu linear -> linear -> sigmoid.
__global__ void k_heads(const float* __restrict__ net1,
                        const float* __restrict__ task,
                        const float* __restrict__ cp1,
                        const float* __restrict__ Wt, const float* __restrict__ bt,
                        const float* __restrict__ Wc, const float* __restrict__ bc,
                        const float* __restrict__ W1, const float* __restrict__ b1,
                        const float* __restrict__ W2, const float* __restrict__ b2,
                        float* __restrict__ out) {
  __shared__ float feat[192];
  __shared__ float h[128];
  int b = blockIdx.x, t = threadIdx.x;
  if (t < 128) feat[t] = net1[b * 128 + t];
  if (t < 32) {
    float s = bt[t];
    for (int c = 0; c < 3; c++) s += task[b * 3 + c] * Wt[t * 3 + c];
    feat[128 + t] = s;
  } else if (t < 64) {
    int j = t - 32;
    float s = bc[j];
    for (int c = 0; c < 3; c++) s += cp1[b * 3 + c] * Wc[j * 3 + c];
    feat[160 + j] = s;
  }
  __syncthreads();
  float s = b1[t];
  for (int k = 0; k < 192; k++) s += feat[k] * W1[t * 192 + k];
  h[t] = (s > 0.0f) ? s : 0.01f * s;
  __syncthreads();
  if (t == 0) {
    float l = b2[0];
    for (int k = 0; k < 128; k++) l += h[k] * W2[k];
    out[b] = 1.0f / (1.0f + expf(-l));
  }
}

// ---------------------------------------------------------------------------
struct LayerP { const float* W; const float* g; const float* b; int O; int K; };

extern "C" void kernel_launch(void* const* d_in, const int* in_sizes, int n_in,
                              void* d_out, int out_size, void* d_ws,
                              size_t ws_size, hipStream_t stream) {
  (void)in_sizes; (void)n_in; (void)out_size; (void)ws_size;
  const int Bn = 16, Nn = 4096, NS = 32;

  // ---- parse inputs (dict order, nested lists flattened depth-first) ----
  int ip = 0;
  auto nextf = [&]() { return (const float*)d_in[ip++]; };
  const float* pcs  = nextf();
  const float* task = nextf();
  const float* cp1  = nextf();

  const int SAD[4][4] = {{6,32,32,64},{67,64,64,128},{131,128,128,256},{259,256,256,512}};
  LayerP sa[4][3];
  for (int m = 0; m < 4; m++)
    for (int l = 0; l < 3; l++) {
      LayerP& L = sa[m][l];
      L.K = SAD[m][l]; L.O = SAD[m][l + 1];
      L.W = nextf(); L.g = nextf(); L.b = nextf();
    }
  const int FPD[4][4] = {{131,128,128,128},{320,256,128,0},{384,256,256,0},{768,256,256,0}};
  const int fpn[4] = {3, 2, 2, 2};
  LayerP fp[4][3];
  for (int m = 0; m < 4; m++)
    for (int l = 0; l < fpn[m]; l++) {
      LayerP& L = fp[m][l];
      L.K = FPD[m][l]; L.O = FPD[m][l + 1];
      L.W = nextf(); L.g = nextf(); L.b = nextf();
    }
  LayerP fc; fc.K = 128; fc.O = 128;
  fc.W = nextf(); fc.g = nextf(); fc.b = nextf();
  const float* Wt = nextf(); const float* bt = nextf();
  const float* Wc = nextf(); const float* bc = nextf();
  const float* W1 = nextf(); const float* b1 = nextf();
  const float* W2 = nextf(); const float* b2 = nextf();

  // ---- workspace layout ----
  char* base = (char*)d_ws; size_t off = 0;
  auto alloc = [&](size_t bytes) -> void* {
    void* p = base + off; off = (off + bytes + 255) & ~(size_t)255; return p;
  };
  float* pcs_mod = (float*)alloc(sizeof(float) * Bn * Nn * 3);
  const int NP[5] = {4096, 1024, 256, 64, 16};
  float* xyzL[5]; xyzL[0] = pcs_mod;
  for (int l = 1; l <= 4; l++) xyzL[l] = (float*)alloc(sizeof(float) * Bn * NP[l] * 3);
  int* fidx = (int*)alloc(sizeof(int) * Bn * 1024);
  int* bidx = (int*)alloc(sizeof(int) * Bn * 1024 * NS);
  int* nidx = (int*)alloc(sizeof(int) * Bn * Nn * 3);
  float* nw = (float*)alloc(sizeof(float) * Bn * Nn * 3);
  const int FD[5] = {3, 64, 128, 256, 512};
  float* feL[5]; feL[0] = pcs_mod;
  for (int l = 1; l <= 4; l++) feL[l] = (float*)alloc(sizeof(float) * Bn * NP[l] * FD[l]);
  float* ge3 = (float*)alloc(sizeof(float) * Bn * 64 * 256);
  float* ge2 = (float*)alloc(sizeof(float) * Bn * 256 * 256);
  float* ge1 = (float*)alloc(sizeof(float) * Bn * 1024 * 128);
  float* ge0 = (float*)alloc(sizeof(float) * Bn * 4096 * 128);
  float* net1 = (float*)alloc(sizeof(float) * Bn * 128);
  float* stats = (float*)alloc(sizeof(float) * 1024);
  float* ssb   = (float*)alloc(sizeof(float) * 1024);
  _Float16* Wh = (_Float16*)alloc(sizeof(_Float16) * 262144);
  _Float16* A0 = (_Float16*)alloc(sizeof(_Float16) * 16777216);
  _Float16* A1 = (_Float16*)alloc(sizeof(_Float16) * 16777216);

  auto zgrid = [](size_t n) { size_t g = (n + 255) / 256; return (int)(g > 8192 ? 8192 : g); };
  auto run_layer = [&](const LayerP& L, int Kpad, const _Float16* Ain, int M,
                       int mode, _Float16* Anext, float* fout, int aux) {
    k_zero_f<<<1, 256, 0, stream>>>(stats, (size_t)(2 * L.O));
    int wt = L.O * Kpad;
    k_wcvt<<<(wt + 255) / 256, 256, 0, stream>>>(L.W, Wh, L.K, Kpad, wt);
    dim3 grid(M / 16, L.O / 16);
    k_gemm_stats<<<grid, 32, 0, stream>>>(Ain, Wh, Kpad, L.O, stats);
    k_finalize<<<1, L.O, 0, stream>>>(stats, L.g, L.b, 1.0f / (float)M, L.O, ssb);
    k_gemm_apply<<<grid, 32, 0, stream>>>(Ain, Wh, Kpad, L.O, ssb, mode, Anext, fout, aux);
  };

  // ---- pcs with row0 := cp1 ----
  {
    int tot = Bn * Nn * 3;
    k_init_pcs<<<(tot + 255) / 256, 256, 0, stream>>>(pcs, cp1, pcs_mod, Nn, tot);
  }

  // ---- Set Abstraction modules ----
  const float RAD[4] = {0.1f, 0.2f, 0.4f, 0.8f};
  for (int lvl = 0; lvl < 4; lvl++) {
    int Nsrc = NP[lvl], np = NP[lvl + 1];
    k_fps<<<Bn, 256, 0, stream>>>(xyzL[lvl], Nsrc, np, fidx);
    {
      int tot = Bn * np * 3;
      k_gather_xyz<<<(tot + 255) / 256, 256, 0, stream>>>(xyzL[lvl], fidx, xyzL[lvl + 1], np, Nsrc, tot);
    }
    {
      int tot = Bn * np;
      k_ball<<<(tot + 255) / 256, 256, 0, stream>>>(xyzL[lvl], xyzL[lvl + 1], Nsrc, np, NS,
                                                    RAD[lvl] * RAD[lvl], bidx, tot);
    }
    int Cin = 3 + FD[lvl];
    int Kpad = ((Cin + 31) / 32) * 32;
    int M = Bn * np * NS;
    {
      int tot = M * Kpad;
      k_group<<<(tot + 255) / 256, 256, 0, stream>>>(xyzL[lvl], feL[lvl], xyzL[lvl + 1], bidx,
                                                     Nsrc, np, NS, FD[lvl], Kpad, A0, tot);
    }
    size_t pooln = (size_t)Bn * np * FD[lvl + 1];
    k_zero_f<<<zgrid(pooln), 256, 0, stream>>>(feL[lvl + 1], pooln);
    run_layer(sa[lvl][0], Kpad, A0, M, 0, A1, nullptr, 0);
    run_layer(sa[lvl][1], sa[lvl][1].K, A1, M, 0, A0, nullptr, 0);
    run_layer(sa[lvl][2], sa[lvl][2].K, A0, M, 1, nullptr, feL[lvl + 1], NS);
  }

  // ---- Feature Propagation modules (i = 4 .. 1) ----
  float* fin1[4]  = {pcs_mod, feL[1], feL[2], feL[3]};
  int    F1a[4]   = {3, 64, 128, 256};
  float* fin2[4]  = {ge1, ge2, ge3, feL[4]};
  int    F2a[4]   = {128, 256, 256, 512};
  float* fouta[4] = {ge0, ge1, ge2, ge3};
  for (int i = 4; i >= 1; i--) {
    int mi = i - 1;
    int N1 = NP[i - 1], N2 = NP[i];
    int F1d = F1a[mi], F2d = F2a[mi];
    {
      int tot = Bn * N1;
      k_3nn<<<(tot + 255) / 256, 256, 0, stream>>>(xyzL[i - 1], xyzL[i], N1, N2, nidx, nw, tot);
    }
    int Cin = F2d + F1d;
    int Kpad = ((Cin + 31) / 32) * 32;
    int M = Bn * N1;
    {
      int tot = M * Kpad;
      k_fp_build<<<(tot + 255) / 256, 256, 0, stream>>>(fin2[mi], fin1[mi], nidx, nw, N1, N2,
                                                        F2d, F1d, Kpad, A0, tot);
    }
    const _Float16* cur = A0;
    for (int l = 0; l < fpn[mi]; l++) {
      const LayerP& L = fp[mi][l];
      int kp = (l == 0) ? Kpad : L.K;
      bool last = (l == fpn[mi] - 1);
      _Float16* nxt = (cur == A0) ? A1 : A0;
      if (!last) run_layer(L, kp, cur, M, 0, nxt, nullptr, 0);
      else       run_layer(L, kp, cur, M, 2, nullptr, fouta[mi], 0);
      cur = nxt;
    }
  }

  // ---- FC + BN (stats over all rows; only point 0 per batch materialized) ----
  {
    size_t n = (size_t)Bn * Nn * 128;
    k_cast_f2h<<<zgrid(n), 256, 0, stream>>>(ge0, A0, n);
    run_layer(fc, 128, A0, Bn * Nn, 3, nullptr, net1, Nn);
  }

  // ---- Heads ----
  k_heads<<<Bn, 128, 0, stream>>>(net1, task, cp1, Wt, bt, Wc, bc, W1, b1, W2, b2,
                                  (float*)d_out);
}
